// QuadraticNonLinearity_58222576665051
// MI455X (gfx1250) — matrix operand order repaired and checked by reading, compiled, not run on hardware
//
#include <hip/hip_runtime.h>

typedef __attribute__((ext_vector_type(16))) __bf16 v16bf;
typedef __attribute__((ext_vector_type(8)))  float  v8f;

#define NSITES    65536
#define OUT0_OFF  0
#define OUT2_OFF  65536
#define OUT4_OFF  1703936        // 65536 + 65536*25
#define FEATS_OFF 7012352        // OUT4_OFF + 65536*81
#define FEATROW   12352          // 64 + 3*4096

// ---------------------------------------------------------------------------
// Combo table: (l, l1, l2) for the 11 allowed couplings, plus flat offsets of
// each combo's CG block inside the cgl/cgr tables (size (2l+1)(2l1+1)(2l2+1)).
// ---------------------------------------------------------------------------
__device__ const int CMB_L [11] = {0,0,0,2,2,2,2,4,4,4,4};
__device__ const int CMB_L1[11] = {0,2,4,0,2,2,4,0,2,2,4};
__device__ const int CMB_L2[11] = {0,2,4,2,2,4,4,4,2,4,4};
__device__ const int CMB_CGO[11]= {0,1,26,107,132,257,482,887,968,1193,1598}; // sum = 2327

__device__ const double FACT[14] = {1.,1.,2.,6.,24.,120.,720.,5040.,40320.,
                                    362880.,3628800.,39916800.,479001600.,6227020800.};

// ------------------------- SU(2) Clebsch-Gordan (f64) ----------------------
__device__ double su2cg(int j1,int m1,int j2,int m2,int j3,int m3){
  if (m3 != m1 + m2) return 0.0;
  int vmin = max(max(-j1 + j2 + m3, -j1 + m1), 0);
  int vmax = min(min(j2 + j3 + m1, j3 - j1 + j2), j3 + m3);
  double C = sqrt((2.0*j3 + 1.0) *
      FACT[j3+j1-j2]*FACT[j3-j1+j2]*FACT[j1+j2-j3]*FACT[j3+m3]*FACT[j3-m3] /
      (FACT[j1+j2+j3+1]*FACT[j1-m1]*FACT[j1+m1]*FACT[j2-m2]*FACT[j2+m2]));
  double S = 0.0;
  for (int v = vmin; v <= vmax; ++v){
    double t = FACT[j2+j3+m1-v]*FACT[j1-m1+v] /
               (FACT[v]*FACT[j3-j1+j2-v]*FACT[j3+m3-v]*FACT[v+j1-j2-m3]);
    S += ((v + j2 + m2) & 1) ? -t : t;
  }
  return C * S;
}

// real->complex change of basis q (times (-i)^l, real for even l)
__device__ void fillQ(int l, double* qr, double* qi){
  int n = 2*l + 1;
  double ph = ((l & 3) == 2) ? -1.0 : 1.0;     // (-i)^l for even l
  const double is2 = 0.70710678118654752440;
  for (int m = -l; m < 0; ++m){
    qr[(l+m)*n + (l-m)] = is2 * ph;            // q[l+m, l+|m|] = 1/sqrt2
    qi[(l+m)*n + (l+m)] = -is2 * ph;           // q[l+m, l-|m|] = -i/sqrt2
  }
  qr[l*n + l] = ph;
  for (int m = 1; m <= l; ++m){
    double sg = (m & 1) ? -1.0 : 1.0;
    qr[(l+m)*n + (l+m)] = sg * is2 * ph;
    qi[(l+m)*n + (l-m)] = sg * is2 * ph;
  }
}

// ---------------------------------------------------------------------------
// Setup kernel: one block per combo, builds cg_r / cg_l (f32) into d_ws,
// reproducing the reference's transpose->reshape->transpose + sqrt2 scaling.
// ---------------------------------------------------------------------------
__global__ void cg_setup_kernel(float* __restrict__ cglOut, float* __restrict__ cgrOut){
  __shared__ double sC[729];
  __shared__ double sW[729];
  __shared__ double q1r[81], q1i[81], q2r[81], q2i[81], q3r[81], q3i[81];
  __shared__ double sPart[128];
  __shared__ double sScale;
  const int tid = threadIdx.x, nth = blockDim.x;
  const int c  = blockIdx.x;
  const int l3 = CMB_L[c], l1 = CMB_L1[c], l2 = CMB_L2[c];
  const int n1 = 2*l1+1, n2 = 2*l2+1, n3 = 2*l3+1;

  for (int i = tid; i < 729; i += nth){ sC[i] = 0.0; sW[i] = 0.0; }
  for (int i = tid; i < 81;  i += nth){ q1r[i]=0; q1i[i]=0; q2r[i]=0; q2i[i]=0; q3r[i]=0; q3i[i]=0; }
  __syncthreads();
  if (tid == 0){ fillQ(l1, q1r, q1i); fillQ(l2, q2r, q2i); fillQ(l3, q3r, q3i); }
  for (int idx = tid; idx < n1*n2; idx += nth){
    int i = idx / n2, k = idx % n2;
    int m1 = i - l1, m2 = k - l2, m3 = m1 + m2;
    if (m3 >= -l3 && m3 <= l3)
      sC[(i*n2 + k)*n3 + (l3 + m3)] = su2cg(l1, m1, l2, m2, l3, m3);
  }
  __syncthreads();

  // out[j,ll,m] = sum_{i,k,nn} Q1[i,j] Q2[k,ll] conj(Q3[nn,m]) C[i,k,nn]  (real part)
  for (int o = tid; o < n1*n2*n3; o += nth){
    int j = o / (n2*n3); int rem = o - j*(n2*n3);
    int ll = rem / n3;   int m = rem - ll*n3;
    double sr = 0.0;
    for (int i = 0; i < n1; ++i){
      double a1r = q1r[i*n1 + j], a1i = q1i[i*n1 + j];
      if (a1r == 0.0 && a1i == 0.0) continue;
      for (int k = 0; k < n2; ++k){
        double a2r = q2r[k*n2 + ll], a2i = q2i[k*n2 + ll];
        if (a2r == 0.0 && a2i == 0.0) continue;
        double ar = a1r*a2r - a1i*a2i;
        double ai = a1r*a2i + a1i*a2r;
        const double* cp = &sC[(i*n2 + k)*n3];
        for (int nn = 0; nn < n3; ++nn){
          double cv = cp[nn];
          if (cv == 0.0) continue;
          double br = q3r[nn*n3 + m], bi = -q3i[nn*n3 + m]; // conj(Q3[nn,m])
          sr += cv*(ar*br - ai*bi);
        }
      }
    }
    sW[o] = sr;
  }
  __syncthreads();
  double acc = 0.0;
  for (int o = tid; o < n1*n2*n3; o += nth) acc += sW[o]*sW[o];
  sPart[tid] = acc;
  __syncthreads();
  if (tid == 0){
    double t = 0.0;
    for (int i = 0; i < nth; ++i) t += sPart[i];
    double sc = (t > 0.0) ? 1.0/sqrt(t) : 0.0;
    if (l1 != l2) sc *= 1.41421356237309504880;
    sScale = sc;
  }
  __syncthreads();
  const int cgo = CMB_CGO[c];
  // cg_r[k,p,q] = w[t%n1, t/n1, k] with t = p*n2+q ; cg_l[k,q,p] = cg_r[k,p,q]
  for (int idx = tid; idx < n3*n1*n2; idx += nth){
    int k = idx / (n1*n2); int t2 = idx - k*(n1*n2);
    int p = t2 / n2;       int q = t2 - p*n2;
    int kk = t2 / n1;      int ii = t2 - kk*n1;
    double val = sW[(ii*n2 + kk)*n3 + k] * sScale;
    cgrOut[cgo + (k*n1 + p)*n2 + q] = (float)val;
    cglOut[cgo + (k*n2 + q)*n1 + p] = (float)val;
  }
}

// ---------------------------------------------------------------------------
// Main kernel helpers
// ---------------------------------------------------------------------------
__device__ __forceinline__ __bf16 tobf(float x){
  union { float f; unsigned u; } cc; cc.f = x;
  unsigned r = cc.u + 0x7FFFu + ((cc.u >> 16) & 1u);      // RNE
  union { unsigned short s; __bf16 b; } o;
  o.s = (unsigned short)(r >> 16);
  return o.b;
}

struct EDec { int n1, n2, p, q, cgo; };

// Map concatenated-K index -> (combo shape, p, q, cg offset) for each output l.
// With fully-unrolled callers K is (almost) a compile-time constant, so the
// branches fold away and n1/n2/cgo become literals.
__device__ __forceinline__ EDec decodeK(int grp, int K){
  if (grp == 4){                                     // combos (4,0,4)(4,2,2)(4,2,4)(4,4,4), Ftot=160
    if (K < 9)  return EDec{1,9,0,K,887};
    if (K < 34){ int f = K-9;  return EDec{5,5,f/5,f%5,968};  }
    if (K < 79){ int f = K-34; return EDec{5,9,f/9,f%9,1193}; }
    { int f = K-79; return EDec{9,9,f/9,f%9,1598}; }
  } else if (grp == 2){                              // combos (2,0,2)(2,2,2)(2,2,4)(2,4,4), Ftot=156
    if (K >= 156) return EDec{0,0,0,0,0};            // K padding -> zero element
    if (K < 5)  return EDec{1,5,0,K,107};
    if (K < 30){ int f = K-5;  return EDec{5,5,f/5,f%5,132}; }
    if (K < 75){ int f = K-30; return EDec{5,9,f/9,f%9,257}; }
    { int f = K-75; return EDec{9,9,f/9,f%9,482}; }
  } else {                                           // combos (0,0,0)(0,2,2)(0,4,4), Ftot=107
    if (K < 1)  return EDec{1,1,0,0,0};
    if (K < 26){ int f = K-1;  return EDec{5,5,f/5,f%5,1}; }
    { int f = K-26; return EDec{9,9,f/9,f%9,26}; }
  }
}

__device__ __forceinline__ const float* rhSel(int nn, const float* r0, const float* r2, const float* r4){
  return (nn == 1) ? r0 : ((nn == 5) ? r2 : r4);
}

// x_l[k,(p,q)] = sum_j rh[l1][j,p] * cg_l[k,q,j]
__device__ __forceinline__ float xlElem(int grp, int K, int k, const float* cgl,
                                        const float* r0, const float* r2, const float* r4){
  EDec d = decodeK(grp, K);
  const float* rh = rhSel(d.n1, r0, r2, r4);
  const float* cg = cgl + d.cgo + (k*d.n2 + d.q)*d.n1;
  float a = 0.f;
#pragma unroll
  for (int j = 0; j < d.n1; ++j) a += rh[j*d.n1 + d.p] * cg[j];
  return a;
}

// x_r[k,(p,q)] = sum_j rh[l2][q,j] * cg_r[k,p,j]
__device__ __forceinline__ float xrElem(int grp, int K, int k, const float* cgr,
                                        const float* r0, const float* r2, const float* r4){
  EDec d = decodeK(grp, K);
  const float* rh = rhSel(d.n2, r0, r2, r4) + d.q*d.n2;
  const float* cg = cgr + d.cgo + (k*d.n1 + d.p)*d.n2;
  float a = 0.f;
#pragma unroll
  for (int j = 0; j < d.n2; ++j) a += rh[j] * cg[j];
  return a;
}

// ---------------------------------------------------------------------------
// Main kernel: 128 threads = 4 waves, 16 sites per wave.
// l=4: 1 site / 16x16 tile, K=160 -> 5x v_wmma_f32_16x16x32_bf16 chain.
// l=2: 3 sites packed per tile (rows/cols 0..14), K=156 (pad to 160).
// l=0: f32 dot (107 terms), split across lane halves + shfl_xor reduce.
// ---------------------------------------------------------------------------
__global__ __launch_bounds__(128) void qnl_main_kernel(
    const float* __restrict__ g_rh0, const float* __restrict__ g_rh2,
    const float* __restrict__ g_rh4, const float* __restrict__ g_cgl,
    const float* __restrict__ g_cgr, float* __restrict__ out)
{
  __shared__ float s_cgl[2327];
  __shared__ float s_cgr[2327];
  __shared__ float s_rh4[4*16*81];
  __shared__ float s_rh2[4*16*25];
  __shared__ float s_rh0[4*16];

  const int tid  = threadIdx.x;
  const int wave = tid >> 5;
  const int lane = tid & 31;
  const int M    = lane & 15;      // WMMA row of A / column of B for this lane
  const bool hi  = lane >= 16;
  const int ha   = hi ? 8 : 0;     // A-matrix half-lane K offset
  const int hb   = hi ? 16 : 0;    // unused: B takes ha, same K order as A

  for (int i = tid; i < 2327; i += 128){ s_cgl[i] = g_cgl[i]; s_cgr[i] = g_cgr[i]; }

  const int siteBase = (blockIdx.x * 4 + wave) * 16;
  float* w4 = s_rh4 + wave*(16*81);
  float* w2 = s_rh2 + wave*(16*25);
  float* w0 = s_rh0 + wave*16;
  for (int i = lane; i < 16*81; i += 32) w4[i] = g_rh4[(size_t)siteBase*81 + i];
  for (int i = lane; i < 16*25; i += 32) w2[i] = g_rh2[(size_t)siteBase*25 + i];
  if (lane < 16) w0[lane] = g_rh0[siteBase + lane];
  __syncthreads();

  // ---------------- l = 4 ----------------
#pragma unroll 1
  for (int s = 0; s < 16; ++s){
    const float* r0 = w0 + s;
    const float* r2 = w2 + s*25;
    const float* r4 = w4 + s*81;
    v8f acc = {0.f,0.f,0.f,0.f,0.f,0.f,0.f,0.f};
    const bool ok = (M < 9);
#pragma unroll
    for (int kc = 0; kc < 5; ++kc){
      v16bf A, B;
#pragma unroll
      for (int e = 0; e < 16; ++e){
        int Ka = kc*32 + ((e & 8) ? 16 : 0) + ha + (e & 7); // A 16x32 layout
        A[e] = tobf(ok ? xlElem(4, Ka, M, s_cgl, r0, r2, r4) : 0.f);
        int Kb = kc*32 + ((e & 8) ? 16 : 0) + ha + (e & 7); // B 32x16 layout
        B[e] = tobf(ok ? xrElem(4, Kb, M, s_cgr, r0, r2, r4) : 0.f);
      }
      acc = __builtin_amdgcn_wmma_f32_16x16x32_bf16(false, A, false, B, (short)0, acc, false, false);
    }
    size_t ob = OUT4_OFF + (size_t)(siteBase + s)*81;
#pragma unroll
    for (int r = 0; r < 8; ++r){
      int Mc = r + (hi ? 8 : 0);
      if (Mc < 9 && M < 9) out[ob + Mc*9 + M] = acc[r];
    }
  }

  // ---------------- l = 2 (3 sites per tile) ----------------
#pragma unroll 1
  for (int sg = 0; sg < 16; sg += 3){
    const int sub  = M / 5;
    const int krow = M - sub*5;
    const int sA   = sg + sub;
    const bool ok  = (M < 15) && (sA < 16);
    const int sU   = ok ? sA : 0;
    const float* r0 = w0 + sU;
    const float* r2 = w2 + sU*25;
    const float* r4 = w4 + sU*81;
    v8f acc = {0.f,0.f,0.f,0.f,0.f,0.f,0.f,0.f};
#pragma unroll
    for (int kc = 0; kc < 5; ++kc){
      v16bf A, B;
#pragma unroll
      for (int e = 0; e < 16; ++e){
        int Ka = kc*32 + ((e & 8) ? 16 : 0) + ha + (e & 7);
        A[e] = tobf(ok ? xlElem(2, Ka, krow, s_cgl, r0, r2, r4) : 0.f);
        int Kb = kc*32 + ((e & 8) ? 16 : 0) + ha + (e & 7);
        B[e] = tobf(ok ? xrElem(2, Kb, krow, s_cgr, r0, r2, r4) : 0.f);
      }
      acc = __builtin_amdgcn_wmma_f32_16x16x32_bf16(false, A, false, B, (short)0, acc, false, false);
    }
#pragma unroll
    for (int r = 0; r < 8; ++r){
      int Mc = r + (hi ? 8 : 0);
      if (Mc < 15 && M < 15){
        int subM = Mc / 5;
        if (subM == sub && (sg + subM) < 16){   // keep only diagonal site blocks
          size_t ob = OUT2_OFF + (size_t)(siteBase + sg + subM)*25;
          out[ob + (Mc - subM*5)*5 + krow] = acc[r];
        }
      }
    }
  }

  // ---------------- l = 0 ----------------
  {
    const int s = M;
    const float* r0 = w0 + s;
    const float* r2 = w2 + s*25;
    const float* r4 = w4 + s*81;
    const int K0 = hi ? 54 : 0;
    const int Kn = hi ? 53 : 54;
    float a = 0.f;
#pragma unroll 1
    for (int t = 0; t < Kn; ++t){
      int K = K0 + t;
      a += xlElem(0, K, 0, s_cgl, r0, r2, r4) * xrElem(0, K, 0, s_cgr, r0, r2, r4);
    }
    a += __shfl_xor(a, 16, 32);
    if (!hi) out[OUT0_OFF + siteBase + s] = a;
  }
}

// ---------------------------------------------------------------------------
// Rotation-invariant features: feats_out = [feats, n_l * ||rh_n[l]||^2 ...]
// ---------------------------------------------------------------------------
__global__ void qnl_feats_kernel(float* __restrict__ out, const float* __restrict__ feats){
  int site = blockIdx.x * blockDim.x + threadIdx.x;
  if (site < NSITES){
    int n = site >> 12, abc = site & 4095;
    float v0 = out[OUT0_OFF + site];
    float s0 = v0*v0;
    float s2 = 0.f;
    const float* p2 = out + OUT2_OFF + (size_t)site*25;
    for (int i = 0; i < 25; ++i) s2 += p2[i]*p2[i];
    float s4 = 0.f;
    const float* p4 = out + OUT4_OFF + (size_t)site*81;
    for (int i = 0; i < 81; ++i) s4 += p4[i]*p4[i];
    float* fo = out + FEATS_OFF + (size_t)n*FEATROW;
    fo[64 + abc]        = 78.95683520871486895f * s0;   // 8*pi^2 / 1
    fo[64 + 4096 + abc] = 15.79136704174297379f * s2;   // 8*pi^2 / 5
    fo[64 + 8192 + abc] =  8.77298168985720766f * s4;   // 8*pi^2 / 9
  }
  if (site < 1024)
    out[FEATS_OFF + (size_t)(site >> 6)*FEATROW + (site & 63)] = feats[site];
}

// ---------------------------------------------------------------------------
extern "C" void kernel_launch(void* const* d_in, const int* in_sizes, int n_in,
                              void* d_out, int out_size, void* d_ws, size_t ws_size,
                              hipStream_t stream){
  const float* rh0   = (const float*)d_in[0];
  const float* rh2   = (const float*)d_in[1];
  const float* rh4   = (const float*)d_in[2];
  const float* feats = (const float*)d_in[3];
  float* out = (float*)d_out;
  float* cgl = (float*)d_ws;          // 2327 f32
  float* cgr = cgl + 2327;            // 2327 f32   (18.6 KB total)

  cg_setup_kernel<<<dim3(11),   dim3(128), 0, stream>>>(cgl, cgr);
  qnl_main_kernel<<<dim3(1024), dim3(128), 0, stream>>>(rh0, rh2, rh4, cgl, cgr, out);
  qnl_feats_kernel<<<dim3(256), dim3(256), 0, stream>>>(out, feats);
}
